// STM_GrootV3D_V2_46978352283915
// MI455X (gfx1250) — matrix operand order repaired and checked by reading, not compiled, not run on hardware
//
#include <hip/hip_runtime.h>
#include <hip/hip_bf16.h>

typedef __attribute__((ext_vector_type(16))) _Float16 v16h;
typedef __attribute__((ext_vector_type(8)))  float    v8f;
typedef _Float16 half_t;

#define EPSV 1e-5f

__device__ __forceinline__ float sigmoidf_(float x) { return 1.f / (1.f + __expf(-x)); }
__device__ __forceinline__ float softplusf_(float x) { return (x > 20.f) ? x : log1pf(__expf(x)); }

// ---------------------------------------------------------------------------
// Weight packing: B[K,N] -> Bp[K/32][N][32]  (f16), zero padded.
// mode 0: src row-major [Ksrc][Nsrc]
// mode 1: src row-major [Nsrc][Ksrc]   (transposed, e.g. conv2_w (O,C))
// mode 2: 3x3 conv weights (O, C, 3, 3): k = t*C + ci, t = ky*3+kx
// ---------------------------------------------------------------------------
__global__ void pack_b_kernel(const float* __restrict__ src, half_t* __restrict__ dst,
                              int Kpad, int Npad, int Ksrc, int Nsrc, int mode, int C)
{
    int idx = blockIdx.x * blockDim.x + threadIdx.x;
    if (idx >= Kpad * Npad) return;
    int ki = idx & 31;
    int rest = idx >> 5;
    int n = rest % Npad;
    int kt = rest / Npad;
    int k = kt * 32 + ki;
    float v = 0.f;
    if (k < Ksrc && n < Nsrc) {
        if (mode == 0)      v = src[(size_t)k * Nsrc + n];
        else if (mode == 1) v = src[(size_t)n * Ksrc + k];
        else                v = src[((size_t)n * C + (k % C)) * 9 + (k / C)];
    }
    dst[((size_t)kt * Npad + n) * 32 + ki] = (half_t)v;
}

// ---------------------------------------------------------------------------
// Build concat(x,y) as f16 matrix Xmat[M=16384][K=512], m=(b*64+y)*128+x'
// ---------------------------------------------------------------------------
__global__ void build_xmat_kernel(const float* __restrict__ xin, const float* __restrict__ yin,
                                  half_t* __restrict__ Xmat)
{
    int idx = blockIdx.x * blockDim.x + threadIdx.x;   // m*512 + c
    if (idx >= 16384 * 512) return;
    int c = idx & 511;
    int m = idx >> 9;
    int b  = m >> 13;
    int yv = (m >> 7) & 63;
    int xp = m & 127;
    const float* src = (xp < 64) ? xin : yin;
    int xx = xp & 63;
    float v = src[(((size_t)(b * 512 + c)) * 64 + yv) * 64 + xx];
    Xmat[(size_t)m * 512 + c] = (half_t)v;
}

// ---------------------------------------------------------------------------
// Generic WMMA GEMM with fused epilogues.
// block = 256 threads (8 waves); wave w -> 32x32 output tile:
//   rows [bx*256 + w*32, +32), cols [by*32, +32)   (2 A frags x 2 B frags)
// Software-pipelined fragment loads (double buffer) so WMMAs only wait on
// loads issued one iteration earlier.
// ---------------------------------------------------------------------------
struct Epi {
    int mode;                 // 0: out32 (+bias) | 2: BN+ReLU -> out16 | 3: BN+resid+ReLU -> NCHW dout | 4: +resid -> out32 & out16
    float*  out32;
    half_t* out16;
    int     ldo;
    const float* bias;
    const float* bng; const float* bnb; const float* bnm; const float* bnv;
    const float* resid;
    float*  dout;
    int     sel;
};

__device__ __forceinline__ void epi_elem(const Epi& ep, int m, int n, float val)
{
    if (ep.mode == 0) {
        if (ep.bias) val += ep.bias[n];
        ep.out32[(size_t)m * ep.ldo + n] = val;
    } else if (ep.mode == 2) {
        float s = ep.bng[n] * rsqrtf(ep.bnv[n] + EPSV);
        val = (val - ep.bnm[n]) * s + ep.bnb[n];
        val = fmaxf(val, 0.f);
        ep.out16[(size_t)m * ep.ldo + n] = (half_t)val;
    } else if (ep.mode == 3) {
        float s = ep.bng[n] * rsqrtf(ep.bnv[n] + EPSV);
        val = (val - ep.bnm[n]) * s + ep.bnb[n];
        int b = m >> 12, y = (m >> 6) & 63, x = m & 63;
        int ridx = (b * 64 + y) * 128 + ep.sel * 64 + x;
        val += ep.resid[(size_t)ridx * 128 + n];
        val = fmaxf(val, 0.f);
        ep.dout[(size_t)ep.sel * 2097152 + ((size_t)(b * 128 + n) << 12) + (y << 6) + x] = val;
    } else { // mode 4: out_proj, += seq residual, dual write
        val += ep.resid[(size_t)m * 128 + n];
        ep.out32[(size_t)m * ep.ldo + n] = val;
        ep.out16[(size_t)m * ep.ldo + n] = (half_t)val;
    }
}

__global__ __launch_bounds__(256)
void wmma_gemm_kernel(const half_t* __restrict__ A, const half_t* __restrict__ Bp,
                      int K, int N, Epi ep)
{
    const int wave = threadIdx.x >> 5;
    const int lane = threadIdx.x & 31;
    const int l15  = lane & 15;
    const int hi   = lane >> 4;
    const int m0   = blockIdx.x * 256 + wave * 32;
    const int n0   = blockIdx.y * 32;

    v8f acc00 = {0.f,0.f,0.f,0.f,0.f,0.f,0.f,0.f};
    v8f acc01 = acc00, acc10 = acc00, acc11 = acc00;

    // per-lane base pointers with K-sub-offsets folded in (contiguous k = 16*hi + s, same order for A and B)
    const half_t* Arow0 = A + (size_t)(m0 + l15) * K + hi * 16;       // rows m0..m0+15
    const half_t* Arow1 = Arow0 + (size_t)16 * K;                     // rows m0+16..m0+31
    const half_t* Bcol0 = Bp + (size_t)(n0 + l15) * 32 + hi * 16;     // cols n0..n0+15
    const half_t* Bcol1 = Bp + (size_t)(n0 + 16 + l15) * 32 + hi * 16;// cols n0+16..n0+31
    const size_t bstride = (size_t)N * 32;

    union U { uint4 u[2]; v16h v; };
    U a0c, a1c; v16h b0c, b1c;

    // preload k = 0
    a0c.u[0] = *(const uint4*)(Arow0);
    a0c.u[1] = *(const uint4*)(Arow0 + 8);
    a1c.u[0] = *(const uint4*)(Arow1);
    a1c.u[1] = *(const uint4*)(Arow1 + 8);
    b0c = *(const v16h*)(Bcol0);
    b1c = *(const v16h*)(Bcol1);

    int k0 = 0;
    for (; k0 < K - 32; k0 += 32) {
        const int k1 = k0 + 32;
        U a0n, a1n; v16h b0n, b1n;
        a0n.u[0] = *(const uint4*)(Arow0 + k1);
        a0n.u[1] = *(const uint4*)(Arow0 + k1 + 8);
        a1n.u[0] = *(const uint4*)(Arow1 + k1);
        a1n.u[1] = *(const uint4*)(Arow1 + k1 + 8);
        const size_t bo = (size_t)(k1 >> 5) * bstride;
        b0n = *(const v16h*)(Bcol0 + bo);
        b1n = *(const v16h*)(Bcol1 + bo);
        __builtin_prefetch(Arow0 + k1 + 64, 0, 3);
        __builtin_prefetch(Arow1 + k1 + 64, 0, 3);

        acc00 = __builtin_amdgcn_wmma_f32_16x16x32_f16(false, a0c.v, false, b0c, (short)0, acc00, false, false);
        acc01 = __builtin_amdgcn_wmma_f32_16x16x32_f16(false, a0c.v, false, b1c, (short)0, acc01, false, false);
        acc10 = __builtin_amdgcn_wmma_f32_16x16x32_f16(false, a1c.v, false, b0c, (short)0, acc10, false, false);
        acc11 = __builtin_amdgcn_wmma_f32_16x16x32_f16(false, a1c.v, false, b1c, (short)0, acc11, false, false);

        a0c = a0n; a1c = a1n; b0c = b0n; b1c = b1n;
    }
    // last K step
    acc00 = __builtin_amdgcn_wmma_f32_16x16x32_f16(false, a0c.v, false, b0c, (short)0, acc00, false, false);
    acc01 = __builtin_amdgcn_wmma_f32_16x16x32_f16(false, a0c.v, false, b1c, (short)0, acc01, false, false);
    acc10 = __builtin_amdgcn_wmma_f32_16x16x32_f16(false, a1c.v, false, b0c, (short)0, acc10, false, false);
    acc11 = __builtin_amdgcn_wmma_f32_16x16x32_f16(false, a1c.v, false, b1c, (short)0, acc11, false, false);

    // epilogue: C/D layout: lane col = tileN + (lane&15); VGPR v -> row = v + 8*(lane>>4)
    #pragma unroll
    for (int v = 0; v < 8; ++v) {
        int mA = m0 + hi * 8 + v;        // rows for mtile 0
        int mB = mA + 16;                // rows for mtile 1
        int nA = n0 + l15;               // cols for ntile 0
        int nB = nA + 16;                // cols for ntile 1
        epi_elem(ep, mA, nA, acc00[v]);
        epi_elem(ep, mA, nB, acc01[v]);
        epi_elem(ep, mB, nA, acc10[v]);
        epi_elem(ep, mB, nB, acc11[v]);
    }
}

// ---------------------------------------------------------------------------
// LayerNorm over C=128: one wave per row, f16 output
// ---------------------------------------------------------------------------
__global__ __launch_bounds__(256)
void layernorm_kernel(const float* __restrict__ seq, const float* __restrict__ g,
                      const float* __restrict__ bta, half_t* __restrict__ hn)
{
    int wave = threadIdx.x >> 5, lane = threadIdx.x & 31;
    int row = blockIdx.x * 8 + wave;
    float4 v = *(const float4*)(seq + (size_t)row * 128 + lane * 4);
    float s = v.x + v.y + v.z + v.w;
    float q = v.x * v.x + v.y * v.y + v.z * v.z + v.w * v.w;
    #pragma unroll
    for (int off = 16; off > 0; off >>= 1) {
        s += __shfl_xor(s, off, 32);
        q += __shfl_xor(q, off, 32);
    }
    float mu  = s * (1.f / 128.f);
    float var = q * (1.f / 128.f) - mu * mu;
    float rs  = rsqrtf(var + EPSV);
    int c = lane * 4;
    hn[(size_t)row * 128 + c + 0] = (half_t)((v.x - mu) * rs * g[c + 0] + bta[c + 0]);
    hn[(size_t)row * 128 + c + 1] = (half_t)((v.y - mu) * rs * g[c + 1] + bta[c + 1]);
    hn[(size_t)row * 128 + c + 2] = (half_t)((v.z - mu) * rs * g[c + 2] + bta[c + 2]);
    hn[(size_t)row * 128 + c + 3] = (half_t)((v.w - mu) * rs * g[c + 3] + bta[c + 3]);
}

// ---------------------------------------------------------------------------
// Depthwise causal conv1d (k=4) + SiLU; reads u = uz[:, :128]
// ---------------------------------------------------------------------------
__global__ void conv1d_silu_kernel(const float* __restrict__ uz, const float* __restrict__ w,
                                   const float* __restrict__ b, float* __restrict__ uf32,
                                   half_t* __restrict__ uf16)
{
    int idx = blockIdx.x * blockDim.x + threadIdx.x;     // l*128 + d
    if (idx >= 16384 * 128) return;
    int d = idx & 127;
    int l = idx >> 7;
    int ll = l & 8191;                                   // position within batch sequence
    float acc = b[d];
    #pragma unroll
    for (int j = 0; j < 4; ++j) {
        int off = j - 3;
        if (ll + off >= 0) acc += w[d * 4 + j] * uz[(size_t)(l + off) * 256 + d];
    }
    float sv = acc * sigmoidf_(acc);
    uf32[idx] = sv;
    uf16[idx] = (half_t)sv;
}

// ---------------------------------------------------------------------------
// dt = softplus(dbl[:, :8] @ dt_proj_w + dt_proj_b)
// ---------------------------------------------------------------------------
__global__ void dt_kernel(const float* __restrict__ dbl, const float* __restrict__ dtw,
                          const float* __restrict__ dtb, float* __restrict__ dt)
{
    int idx = blockIdx.x * blockDim.x + threadIdx.x;     // l*128 + d
    if (idx >= 16384 * 128) return;
    int d = idx & 127;
    int l = idx >> 7;
    float x = dtb[d];
    #pragma unroll
    for (int r = 0; r < 8; ++r) x += dbl[(size_t)l * 64 + r] * dtw[r * 128 + d];
    dt[idx] = softplusf_(x);
}

// ---------------------------------------------------------------------------
// Chunked selective scan. tid = ((b*128+d)*32 + chunk)*16 + s, chunk len 256.
// ---------------------------------------------------------------------------
__global__ __launch_bounds__(256)
void scan_chunk_kernel(const float* __restrict__ dt, const float* __restrict__ uf,
                       const float* __restrict__ dbl, const float* __restrict__ A_log,
                       float* __restrict__ aggA, float* __restrict__ aggB)
{
    int tid = blockIdx.x * blockDim.x + threadIdx.x;     // 131072
    int s     = tid & 15;
    int chunk = (tid >> 4) & 31;
    int d     = (tid >> 9) & 127;
    int b     = tid >> 16;
    float Aneg = -__expf(A_log[d * 16 + s]);
    float Ap = 1.f, Bg = 0.f;
    int l0 = b * 8192 + chunk * 256;
    for (int i = 0; i < 256; ++i) {
        int l = l0 + i;
        float dtv = dt[(size_t)l * 128 + d];
        float uv  = uf[(size_t)l * 128 + d];
        float Bmv = dbl[(size_t)l * 64 + 8 + s];
        float ab  = __expf(dtv * Aneg);
        Ap = ab * Ap;
        Bg = ab * Bg + dtv * uv * Bmv;
    }
    aggA[tid] = Ap;
    aggB[tid] = Bg;
}

__global__ void scan_combine_kernel(const float* __restrict__ aggA, const float* __restrict__ aggB,
                                    float* __restrict__ carry)
{
    int tid = blockIdx.x * blockDim.x + threadIdx.x;     // 4096
    if (tid >= 4096) return;
    int s = tid & 15;
    int d = (tid >> 4) & 127;
    int b = tid >> 11;
    float h = 0.f;
    for (int c = 0; c < 32; ++c) {
        int idx = (((b * 128 + d) * 32) + c) * 16 + s;
        carry[idx] = h;
        h = aggA[idx] * h + aggB[idx];
    }
}

__global__ __launch_bounds__(256)
void scan_final_kernel(const float* __restrict__ dt, const float* __restrict__ uf,
                       const float* __restrict__ dbl, const float* __restrict__ A_log,
                       const float* __restrict__ Dskip, const float* __restrict__ carry,
                       float* __restrict__ ys)
{
    int tid = blockIdx.x * blockDim.x + threadIdx.x;     // 131072
    int s     = tid & 15;
    int chunk = (tid >> 4) & 31;
    int d     = (tid >> 9) & 127;
    int b     = tid >> 16;
    float Aneg = -__expf(A_log[d * 16 + s]);
    float h    = carry[tid];
    float dsk  = Dskip[d];
    int l0 = b * 8192 + chunk * 256;
    for (int i = 0; i < 256; ++i) {
        int l = l0 + i;
        float dtv = dt[(size_t)l * 128 + d];
        float uv  = uf[(size_t)l * 128 + d];
        float Bmv = dbl[(size_t)l * 64 + 8 + s];
        float Cmv = dbl[(size_t)l * 64 + 24 + s];
        float ab  = __expf(dtv * Aneg);
        h = ab * h + dtv * uv * Bmv;
        float yv = h * Cmv;
        yv += __shfl_xor(yv, 8, 16);
        yv += __shfl_xor(yv, 4, 16);
        yv += __shfl_xor(yv, 2, 16);
        yv += __shfl_xor(yv, 1, 16);
        if (s == 0) ys[(size_t)l * 128 + d] = yv + uv * dsk;
    }
}

// ---------------------------------------------------------------------------
// gated = ys * silu(z)
// ---------------------------------------------------------------------------
__global__ void gate_kernel(const float* __restrict__ ys, const float* __restrict__ uz,
                            half_t* __restrict__ gated)
{
    int idx = blockIdx.x * blockDim.x + threadIdx.x;     // l*128 + d
    if (idx >= 16384 * 128) return;
    int d = idx & 127;
    int l = idx >> 7;
    float z = uz[(size_t)l * 256 + 128 + d];
    gated[idx] = (half_t)(ys[idx] * z * sigmoidf_(z));
}

// ---------------------------------------------------------------------------
// im2col for 3x3 pad-1 conv on NHWC f16 src (image width rowW, x offset xoff)
// dst[M=8192][K=1152], K ordered (ky,kx,c). One thread per 8 channels.
// ---------------------------------------------------------------------------
__global__ void im2col_kernel(const half_t* __restrict__ src, int rowW, int xoff,
                              half_t* __restrict__ dst)
{
    int idx = blockIdx.x * blockDim.x + threadIdx.x;     // 8192*9*16
    if (idx >= 8192 * 9 * 16) return;
    int c8 = idx & 15;
    int t  = (idx >> 4) % 9;
    int m  = idx / 144;
    int b = m >> 12, y = (m >> 6) & 63, x = m & 63;
    int ky = t / 3, kx = t % 3;
    int yy = y + ky - 1, xx = x + kx - 1;
    uint4 val = {0u, 0u, 0u, 0u};
    if ((unsigned)yy < 64u && (unsigned)xx < 64u)
        val = *(const uint4*)(src + ((size_t)((b * 64 + yy) * rowW + xoff + xx)) * 128 + c8 * 8);
    *(uint4*)(dst + (size_t)m * 1152 + t * 128 + c8 * 8) = val;
}

// ---------------------------------------------------------------------------
// Host-side orchestration
// ---------------------------------------------------------------------------
static inline Epi epi_zero() {
    Epi e; e.mode = 0; e.out32 = nullptr; e.out16 = nullptr; e.ldo = 0;
    e.bias = nullptr; e.bng = nullptr; e.bnb = nullptr; e.bnm = nullptr; e.bnv = nullptr;
    e.resid = nullptr; e.dout = nullptr; e.sel = 0; return e;
}

extern "C" void kernel_launch(void* const* d_in, const int* in_sizes, int n_in,
                              void* d_out, int out_size, void* d_ws, size_t ws_size,
                              hipStream_t stream)
{
    (void)in_sizes; (void)n_in; (void)out_size; (void)ws_size;

    const float* xin       = (const float*)d_in[0];
    const float* yin       = (const float*)d_in[1];
    const float* conv2_w   = (const float*)d_in[2];
    const float* conv2_b   = (const float*)d_in[3];
    const float* norm_g    = (const float*)d_in[4];
    const float* norm_b    = (const float*)d_in[5];
    const float* in_proj_w = (const float*)d_in[6];
    const float* conv1d_w  = (const float*)d_in[7];
    const float* conv1d_b  = (const float*)d_in[8];
    const float* x_proj_w  = (const float*)d_in[9];
    const float* dt_proj_w = (const float*)d_in[10];
    const float* dt_proj_b = (const float*)d_in[11];
    const float* A_log     = (const float*)d_in[12];
    const float* Dskip     = (const float*)d_in[13];
    const float* out_proj_w= (const float*)d_in[14];
    const float* rb_c1_w   = (const float*)d_in[15];
    const float* rb_bn1_g  = (const float*)d_in[16];
    const float* rb_bn1_b  = (const float*)d_in[17];
    const float* rb_bn1_m  = (const float*)d_in[18];
    const float* rb_bn1_v  = (const float*)d_in[19];
    const float* rb_c2_w   = (const float*)d_in[20];
    const float* rb_bn2_g  = (const float*)d_in[21];
    const float* rb_bn2_b  = (const float*)d_in[22];
    const float* rb_bn2_m  = (const float*)d_in[23];
    const float* rb_bn2_v  = (const float*)d_in[24];

    char* ws = (char*)d_ws;
    // packed weights
    half_t* W1p  = (half_t*)(ws + 0);          // 512x128
    half_t* INPp = (half_t*)(ws + 131072);     // 128x256
    half_t* XPp  = (half_t*)(ws + 196608);     // 128x64 (N padded from 40)
    half_t* OPp  = (half_t*)(ws + 212992);     // 128x128
    half_t* RB1p = (half_t*)(ws + 245760);     // 1152x128
    half_t* RB2p = (half_t*)(ws + 540672);     // 1152x128
    // activations (lifetime-based reuse)
    float*  SEQ    = (float*) (ws + 1048576);   // 16384x128 f32
    float*  GOUT   = (float*) (ws + 9437184);   // 16384x128 f32
    half_t* GOUT16 = (half_t*)(ws + 17825792);  // 16384x128 f16
    half_t* R1     = (half_t*)(ws + 22020096);  // 8192x128 f16
    float*  AGGA   = (float*) (ws + 24117248);  // 131072 f32
    float*  AGGB   = (float*) (ws + 24641536);
    float*  CARRY  = (float*) (ws + 25165824);
    char*   BIG    = ws + 25690112;             // 18,874,368 B: Xmat -> UZ -> Acol
    half_t* XMAT   = (half_t*)BIG;              // 16384x512 f16
    float*  UZ     = (float*) BIG;              // 16384x256 f32
    half_t* ACOL   = (half_t*)BIG;              // 8192x1152 f16
    float*  UF32   = (float*) (ws + 44564480);  // 16384x128 f32
    half_t* HN     = (half_t*)(ws + 52953088);  // 16384x128 f16 (reused as UF16)
    half_t* UF16   = HN;
    float*  DBL    = (float*) (ws + 57147392);  // 16384x64 f32 (reused as GATED f16)
    half_t* GATED  = (half_t*)(ws + 57147392);
    float*  DT     = (float*) (ws + 61341696);  // 16384x128 f32
    float*  YS     = (float*) (ws + 69730304);  // 16384x128 f32
    float*  OUT    = (float*)d_out;

    const int TB = 256;

    // --- pack weights ---
    pack_b_kernel<<<(512*128)/TB, TB, 0, stream>>>(conv2_w,   W1p,  512, 128, 512, 128, 1, 0);
    pack_b_kernel<<<(128*256)/TB, TB, 0, stream>>>(in_proj_w, INPp, 128, 256, 128, 256, 0, 0);
    pack_b_kernel<<<(128*64 )/TB, TB, 0, stream>>>(x_proj_w,  XPp,  128,  64, 128,  40, 0, 0);
    pack_b_kernel<<<(128*128)/TB, TB, 0, stream>>>(out_proj_w,OPp,  128, 128, 128, 128, 0, 0);
    pack_b_kernel<<<(1152*128)/TB, TB, 0, stream>>>(rb_c1_w,  RB1p, 1152,128, 1152,128, 2, 128);
    pack_b_kernel<<<(1152*128)/TB, TB, 0, stream>>>(rb_c2_w,  RB2p, 1152,128, 1152,128, 2, 128);

    // --- concat + f16 conversion ---
    build_xmat_kernel<<<(16384*512)/TB, TB, 0, stream>>>(xin, yin, XMAT);

    // --- 1x1 conv GEMM: SEQ = XMAT @ W1 + conv2_b ---
    { Epi e = epi_zero(); e.mode = 0; e.out32 = SEQ; e.ldo = 128; e.bias = conv2_b;
      wmma_gemm_kernel<<<dim3(64, 4), TB, 0, stream>>>(XMAT, W1p, 512, 128, e); }

    // --- LayerNorm -> HN (f16) ---
    layernorm_kernel<<<2048, TB, 0, stream>>>(SEQ, norm_g, norm_b, HN);

    // --- in_proj GEMM: UZ = HN @ in_proj_w ---
    { Epi e = epi_zero(); e.mode = 0; e.out32 = UZ; e.ldo = 256;
      wmma_gemm_kernel<<<dim3(64, 8), TB, 0, stream>>>(HN, INPp, 128, 256, e); }

    // --- causal conv1d + SiLU -> UF32 / UF16 ---
    conv1d_silu_kernel<<<(16384*128)/TB, TB, 0, stream>>>(UZ, conv1d_w, conv1d_b, UF32, UF16);

    // --- x_proj GEMM: DBL = UF16 @ x_proj_w (N padded to 64) ---
    { Epi e = epi_zero(); e.mode = 0; e.out32 = DBL; e.ldo = 64;
      wmma_gemm_kernel<<<dim3(64, 2), TB, 0, stream>>>(UF16, XPp, 128, 64, e); }

    // --- dt = softplus(dbl[:, :8] @ dt_proj_w + b) ---
    dt_kernel<<<(16384*128)/TB, TB, 0, stream>>>(DBL, dt_proj_w, dt_proj_b, DT);

    // --- chunked selective scan ---
    scan_chunk_kernel  <<<512, TB, 0, stream>>>(DT, UF32, DBL, A_log, AGGA, AGGB);
    scan_combine_kernel<<<16,  TB, 0, stream>>>(AGGA, AGGB, CARRY);
    scan_final_kernel  <<<512, TB, 0, stream>>>(DT, UF32, DBL, A_log, Dskip, CARRY, YS);

    // --- gating (GATED reuses DBL region; DBL dead after scan) ---
    gate_kernel<<<(16384*128)/TB, TB, 0, stream>>>(YS, UZ, GATED);

    // --- out_proj GEMM + seq residual -> GOUT (f32) & GOUT16 (f16) ---
    { Epi e = epi_zero(); e.mode = 4; e.out32 = GOUT; e.out16 = GOUT16; e.ldo = 128; e.resid = SEQ;
      wmma_gemm_kernel<<<dim3(64, 4), TB, 0, stream>>>(GATED, OPp, 128, 128, e); }

    // --- two ResBlocks (xf: sel=0, yf: sel=1), implicit-GEMM 3x3 convs ---
    for (int sel = 0; sel < 2; ++sel) {
        im2col_kernel<<<(8192*9*16)/TB, TB, 0, stream>>>(GOUT16, 128, sel * 64, ACOL);
        { Epi e = epi_zero(); e.mode = 2; e.out16 = R1; e.ldo = 128;
          e.bng = rb_bn1_g; e.bnb = rb_bn1_b; e.bnm = rb_bn1_m; e.bnv = rb_bn1_v;
          wmma_gemm_kernel<<<dim3(32, 4), TB, 0, stream>>>(ACOL, RB1p, 1152, 128, e); }
        im2col_kernel<<<(8192*9*16)/TB, TB, 0, stream>>>(R1, 64, 0, ACOL);
        { Epi e = epi_zero(); e.mode = 3; e.dout = OUT; e.sel = sel; e.resid = GOUT;
          e.bng = rb_bn2_g; e.bnb = rb_bn2_b; e.bnm = rb_bn2_m; e.bnv = rb_bn2_v;
          wmma_gemm_kernel<<<dim3(32, 4), TB, 0, stream>>>(ACOL, RB2p, 1152, 128, e); }
    }
}
